// Gemma4VisionAttention_83562883711240
// MI455X (gfx1250) — matrix operand order repaired and checked by reading, not compiled, not run on hardware
//
#include <hip/hip_runtime.h>

typedef __bf16 bf16_t;
typedef __attribute__((ext_vector_type(4)))  __bf16 v4bf;
typedef __attribute__((ext_vector_type(8)))  __bf16 v8bf;
typedef __attribute__((ext_vector_type(16))) __bf16 v16bf;
typedef __attribute__((ext_vector_type(8)))  float  v8f;
typedef __attribute__((ext_vector_type(4)))  int    i32x4;

#define HIDN 1152
#define NH   16
#define DH   72
#define SEQ  1024
#define DPAD 96      // K-dim padding for q/k (3 chunks of 32)
#define VPAD 80      // N-dim padding for v (5 tiles of 16)

#if defined(__HIP_DEVICE_COMPILE__) && \
    __has_builtin(__builtin_amdgcn_global_load_async_to_lds_b128) && \
    __has_builtin(__builtin_amdgcn_s_wait_asynccnt)
#define USE_ASYNC_LDS 1
#else
#define USE_ASYNC_LDS 0
#endif

#if USE_ASYNC_LDS
__device__ __forceinline__ void async_cp16(void* lds, const void* g) {
  __builtin_amdgcn_global_load_async_to_lds_b128(
      (__attribute__((address_space(1))) i32x4*)g,
      (__attribute__((address_space(3))) i32x4*)lds, 0, 0);
}
#endif

// ---------------------------------------------------------------------------
// fp32 -> bf16 cast, 4 elements/thread (grid-stride)
// ---------------------------------------------------------------------------
__global__ __launch_bounds__(256)
void cast_f32_bf16(const float4* __restrict__ src, bf16_t* __restrict__ dst, int n4) {
  int i = blockIdx.x * blockDim.x + threadIdx.x;
  int stride = gridDim.x * blockDim.x;
  for (; i < n4; i += stride) {
    float4 f = src[i];
    v4bf o = {(bf16_t)f.x, (bf16_t)f.y, (bf16_t)f.z, (bf16_t)f.w};
    *(v4bf*)(dst + 4 * (size_t)i) = o;
  }
}

// ---------------------------------------------------------------------------
// C[M,N] = A[M,K] * B[N,K]^T + bias   (A,B bf16 row-major, C f32)
// 64x64 block tile, BK=32, 8 waves: wave (wr,wc) owns a 16x32 sub-tile.
// Double-buffered LDS; async global->LDS copies when available.
// ---------------------------------------------------------------------------
__global__ __launch_bounds__(256)
void gemm_bf16_wmma(const bf16_t* __restrict__ A, const bf16_t* __restrict__ B,
                    const float* __restrict__ bias, float* __restrict__ C,
                    int M, int N, int K) {
  __shared__ bf16_t As[2][64][32];
  __shared__ bf16_t Bs[2][64][32];
  const int tid  = threadIdx.x;
  const int wave = tid >> 5, lane = tid & 31;
  const int wr = wave >> 1, wc = wave & 1;
  const int hlf = lane >> 4, col = lane & 15;
  const int rowblock = blockIdx.y * 64, colblock = blockIdx.x * 64;
  const int ldrow = tid >> 2;          // 0..63
  const int ldcol = (tid & 3) * 8;     // 0,8,16,24

  const bf16_t* gA = A + (size_t)(rowblock + ldrow) * K + ldcol;
  const bf16_t* gB = B + (size_t)(colblock + ldrow) * K + ldcol;
  const int nt = K >> 5;               // K/32 tiles

  // ---- prologue: tile 0 into buffer 0 ----
#if USE_ASYNC_LDS
  async_cp16(&As[0][ldrow][ldcol], gA);
  async_cp16(&Bs[0][ldrow][ldcol], gB);
  __builtin_amdgcn_s_wait_asynccnt(0);
  __syncthreads();
#else
  *(uint4*)&As[0][ldrow][ldcol] = *(const uint4*)gA;
  *(uint4*)&Bs[0][ldrow][ldcol] = *(const uint4*)gB;
  __syncthreads();
#endif

  v8f acc0 = {}; v8f acc1 = {};
  for (int t = 0; t < nt; ++t) {
    const int cur = t & 1, nxt = cur ^ 1;
    // ---- issue next tile while current tile computes ----
#if USE_ASYNC_LDS
    if (t + 1 < nt) {
      async_cp16(&As[nxt][ldrow][ldcol], gA + (size_t)(t + 1) * 32);
      async_cp16(&Bs[nxt][ldrow][ldcol], gB + (size_t)(t + 1) * 32);
    }
#else
    uint4 na = {0, 0, 0, 0}, nb = {0, 0, 0, 0};
    if (t + 1 < nt) {
      na = *(const uint4*)(gA + (size_t)(t + 1) * 32);
      nb = *(const uint4*)(gB + (size_t)(t + 1) * 32);
    }
#endif
    // ---- compute from current buffer ----
    const int ar = wr * 16 + col;
    v8bf alo = *(const v8bf*)&As[cur][ar][hlf * 16];
    v8bf ahi = *(const v8bf*)&As[cur][ar][hlf * 16 + 8];
    v16bf af = __builtin_shufflevector(alo, ahi,
        0,1,2,3,4,5,6,7,8,9,10,11,12,13,14,15);
    v16bf b0 = *(const v16bf*)&Bs[cur][wc * 32 + col][hlf * 16];
    v16bf b1 = *(const v16bf*)&Bs[cur][wc * 32 + 16 + col][hlf * 16];
    acc0 = __builtin_amdgcn_wmma_f32_16x16x32_bf16(false, af, false, b0,
                                                   (short)0, acc0, false, false);
    acc1 = __builtin_amdgcn_wmma_f32_16x16x32_bf16(false, af, false, b1,
                                                   (short)0, acc1, false, false);
    // ---- publish next buffer; single barrier per iteration ----
#if USE_ASYNC_LDS
    __builtin_amdgcn_s_wait_asynccnt(0);
    __syncthreads();
#else
    if (t + 1 < nt) {
      *(uint4*)&As[nxt][ldrow][ldcol] = na;
      *(uint4*)&Bs[nxt][ldrow][ldcol] = nb;
    }
    __syncthreads();
#endif
  }
#pragma unroll
  for (int v = 0; v < 8; ++v) {
    int m  = rowblock + wr * 16 + v + 8 * hlf;
    int n0 = colblock + wc * 32 + col;
    int n1 = n0 + 16;
    C[(size_t)m * N + n0] = acc0[v] + bias[n0];
    C[(size_t)m * N + n1] = acc1[v] + bias[n1];
  }
}

// ---------------------------------------------------------------------------
// RMSNorm + per-head scale + 2D RoPE, writes:
//   qpad/kpad: [b*H][S][DPAD] bf16 (zero-padded d 72..95)
//   vtr:       [b*H][VPAD][S] bf16 (d-major for PV B-fragments)
// One thread per (b*s, h).
// ---------------------------------------------------------------------------
__global__ __launch_bounds__(256)
void norm_rope(const float* __restrict__ qkv, const float* __restrict__ cosv,
               const float* __restrict__ sinv, const float* __restrict__ q_scale,
               const float* __restrict__ k_scale, bf16_t* __restrict__ qpad,
               bf16_t* __restrict__ kpad, bf16_t* __restrict__ vtr) {
  int gid = blockIdx.x * blockDim.x + threadIdx.x;   // 0..65535
  int bs = gid >> 4;
  int h  = gid & 15;
  int b  = bs >> 10, s = bs & 1023;
  int bh = b * NH + h;
  const float* base = qkv + (size_t)bs * (3 * NH * DH);
  const float* cr = cosv + (size_t)bs * DH;
  const float* sr = sinv + (size_t)bs * DH;

  // --- Q ---
  {
    const float* x = base + h * DH;
    float ss = 0.f;
    for (int d = 0; d < DH; ++d) ss += x[d] * x[d];
    float inv = rsqrtf(ss * (1.0f / DH) + 1e-6f);
    size_t qi = ((size_t)bh * SEQ + s) * DPAD;
    for (int d = 0; d < DH; ++d) {
      int hf = (d >= 36) ? 1 : 0;
      int j  = d - 36 * hf, bb = 36 * hf;
      float xd = x[d] * inv * (1.f + q_scale[d]);
      int  pj  = (j < 18) ? (bb + j + 18) : (bb + j - 18);
      float xp = x[pj] * inv * (1.f + q_scale[pj]);
      float rot = (j < 18) ? -xp : xp;
      qpad[qi + d] = (bf16_t)(xd * cr[d] + rot * sr[d]);
    }
    for (int d = DH; d < DPAD; ++d) qpad[qi + d] = (bf16_t)0.f;
  }
  // --- K ---
  {
    const float* x = base + NH * DH + h * DH;
    float ss = 0.f;
    for (int d = 0; d < DH; ++d) ss += x[d] * x[d];
    float inv = rsqrtf(ss * (1.0f / DH) + 1e-6f);
    size_t ki = ((size_t)bh * SEQ + s) * DPAD;
    for (int d = 0; d < DH; ++d) {
      int hf = (d >= 36) ? 1 : 0;
      int j  = d - 36 * hf, bb = 36 * hf;
      float xd = x[d] * inv * (1.f + k_scale[d]);
      int  pj  = (j < 18) ? (bb + j + 18) : (bb + j - 18);
      float xp = x[pj] * inv * (1.f + k_scale[pj]);
      float rot = (j < 18) ? -xp : xp;
      kpad[ki + d] = (bf16_t)(xd * cr[d] + rot * sr[d]);
    }
    for (int d = DH; d < DPAD; ++d) kpad[ki + d] = (bf16_t)0.f;
  }
  // --- V ---
  {
    const float* x = base + 2 * NH * DH + h * DH;
    float ss = 0.f;
    for (int d = 0; d < DH; ++d) ss += x[d] * x[d];
    float inv = rsqrtf(ss * (1.0f / DH) + 1e-6f);
    size_t vb = (size_t)bh * VPAD * SEQ + s;
    for (int d = 0; d < DH; ++d)
      vtr[vb + (size_t)d * SEQ] = (bf16_t)(x[d] * inv);
    for (int d = DH; d < VPAD; ++d)
      vtr[vb + (size_t)d * SEQ] = (bf16_t)0.f;
  }
}

// ---------------------------------------------------------------------------
// Flash attention: grid (S/64, b*H), 4 waves/block, each wave owns 16 q rows.
// Online softmax merged over 32-key steps; P re-layout via per-wave LDS.
// ---------------------------------------------------------------------------
__global__ __launch_bounds__(128)
void attn_kernel(const bf16_t* __restrict__ qp, const bf16_t* __restrict__ kp,
                 const bf16_t* __restrict__ vt, bf16_t* __restrict__ ao) {
  __shared__ bf16_t P[4][16][32];
  const int tid  = threadIdx.x;
  const int wave = tid >> 5, lane = tid & 31;
  const int hlf = lane >> 4, col = lane & 15;
  const int bh = blockIdx.y;
  const int b = bh >> 4, h = bh & 15;
  const int qbase = blockIdx.x * 64 + wave * 16;

  // Q tile (16 rows x 96) as 3 A-fragments, kept in registers
  const bf16_t* qrow = qp + ((size_t)bh * SEQ + qbase + col) * DPAD;
  v16bf qf[3];
#pragma unroll
  for (int c = 0; c < 3; ++c) {
    v8bf lo = *(const v8bf*)(qrow + c * 32 + hlf * 16);
    v8bf hi = *(const v8bf*)(qrow + c * 32 + hlf * 16 + 8);
    qf[c] = __builtin_shufflevector(lo, hi,
        0,1,2,3,4,5,6,7,8,9,10,11,12,13,14,15);
  }

  v8f O[5] = {};                     // 16 x 80 accumulator
  float m[8], l[8];
#pragma unroll
  for (int v = 0; v < 8; ++v) { m[v] = -1e30f; l[v] = 0.f; }

  const bf16_t* kb = kp + (size_t)bh * SEQ * DPAD;
  const bf16_t* vb = vt + (size_t)bh * VPAD * SEQ;

  for (int kt = 0; kt < SEQ; kt += 32) {
    // prefetch next K tile (lowers to global_prefetch_b8)
    if (kt + 32 < SEQ) {
      __builtin_prefetch(kb + (size_t)(kt + 32 + col) * DPAD, 0, 3);
      __builtin_prefetch(kb + (size_t)(kt + 48 + col) * DPAD, 0, 3);
    }
    // S(16q x 32k) = Q * K^T over d (3 chunks of 32, zero-padded)
    const bf16_t* krow0 = kb + (size_t)(kt + col) * DPAD + hlf * 16;
    const bf16_t* krow1 = kb + (size_t)(kt + 16 + col) * DPAD + hlf * 16;
    v8f S0 = {}, S1 = {};
#pragma unroll
    for (int c = 0; c < 3; ++c) {
      v16bf kf0 = *(const v16bf*)(krow0 + c * 32);
      S0 = __builtin_amdgcn_wmma_f32_16x16x32_bf16(false, qf[c], false, kf0,
                                                   (short)0, S0, false, false);
    }
#pragma unroll
    for (int c = 0; c < 3; ++c) {
      v16bf kf1 = *(const v16bf*)(krow1 + c * 32);
      S1 = __builtin_amdgcn_wmma_f32_16x16x32_bf16(false, qf[c], false, kf1,
                                                   (short)0, S1, false, false);
    }
    // single online-softmax update over all 32 columns
    float sc[8];
#pragma unroll
    for (int v = 0; v < 8; ++v) {
      float x0 = S0[v], x1 = S1[v];
      float rm = fmaxf(x0, x1);
      rm = fmaxf(rm, __shfl_xor(rm, 1, 32));
      rm = fmaxf(rm, __shfl_xor(rm, 2, 32));
      rm = fmaxf(rm, __shfl_xor(rm, 4, 32));
      rm = fmaxf(rm, __shfl_xor(rm, 8, 32));
      float mn = fmaxf(m[v], rm);
      float p0 = __expf(x0 - mn);
      float p1 = __expf(x1 - mn);
      float rs = p0 + p1;
      rs += __shfl_xor(rs, 1, 32);
      rs += __shfl_xor(rs, 2, 32);
      rs += __shfl_xor(rs, 4, 32);
      rs += __shfl_xor(rs, 8, 32);
      sc[v] = __expf(m[v] - mn);
      l[v]  = l[v] * sc[v] + rs;
      m[v]  = mn;
      P[wave][v + 8 * hlf][col]      = (bf16_t)p0;   // C-layout -> LDS
      P[wave][v + 8 * hlf][16 + col] = (bf16_t)p1;
    }
#pragma unroll
    for (int f = 0; f < 5; ++f)
#pragma unroll
      for (int v = 0; v < 8; ++v) O[f][v] *= sc[v];

    // reload P(16x32) as an A-fragment (per-wave LDS, same-wave DS ordering)
    v8bf plo = *(const v8bf*)&P[wave][col][hlf * 16];
    v8bf phi = *(const v8bf*)&P[wave][col][hlf * 16 + 8];
    v16bf pf = __builtin_shufflevector(plo, phi,
        0,1,2,3,4,5,6,7,8,9,10,11,12,13,14,15);
    // O += P * V  (V d-major: contiguous 16 keys per lane column)
#pragma unroll
    for (int f = 0; f < 5; ++f) {
      v16bf vf = *(const v16bf*)(vb + (size_t)(f * 16 + col) * SEQ + kt + hlf * 16);
      O[f] = __builtin_amdgcn_wmma_f32_16x16x32_bf16(false, pf, false, vf,
                                                     (short)0, O[f], false, false);
    }
  }
  // normalize and write [b][s][h*72+d] (bf16 input for output projection)
#pragma unroll
  for (int f = 0; f < 5; ++f) {
    int d = f * 16 + col;
    if (d < DH) {
#pragma unroll
      for (int v = 0; v < 8; ++v) {
        int qr = qbase + v + 8 * hlf;
        ao[((size_t)b * SEQ + qr) * HIDN + h * DH + d] = (bf16_t)(O[f][v] / l[v]);
      }
    }
  }
}

// ---------------------------------------------------------------------------
// Launch
// ---------------------------------------------------------------------------
extern "C" void kernel_launch(void* const* d_in, const int* in_sizes, int n_in,
                              void* d_out, int out_size, void* d_ws, size_t ws_size,
                              hipStream_t stream) {
  (void)in_sizes; (void)n_in; (void)out_size; (void)ws_size;
  const float* hidden  = (const float*)d_in[0];
  const float* cosv    = (const float*)d_in[1];
  const float* sinv    = (const float*)d_in[2];
  const float* w_qkv   = (const float*)d_in[3];
  const float* b_qkv   = (const float*)d_in[4];
  const float* w_o     = (const float*)d_in[5];
  const float* b_o     = (const float*)d_in[6];
  const float* q_scale = (const float*)d_in[7];
  const float* k_scale = (const float*)d_in[8];
  float* out = (float*)d_out;

  char* ws = (char*)d_ws;
  bf16_t* hidden_bf = (bf16_t*)(ws);                 //  9,437,184 B
  bf16_t* wqkv_bf   = (bf16_t*)(ws + 9437184);       //  7,962,624 B
  bf16_t* wo_bf     = (bf16_t*)(ws + 17399808);      //  2,654,208 B
  float*  qkv       = (float*) (ws + 20054016);      // 56,623,104 B
  bf16_t* qpad      = (bf16_t*)(ws + 76677120);      // 12,582,912 B
  bf16_t* kpad      = (bf16_t*)(ws + 89260032);      // 12,582,912 B
  bf16_t* vtr       = (bf16_t*)(ws + 101842944);     // 10,485,760 B
  bf16_t* aout      = (bf16_t*)(ws + 112328704);     //  9,437,184 B

  cast_f32_bf16<<<288, 256, 0, stream>>>((const float4*)hidden, hidden_bf,
                                         4096 * HIDN / 4);
  cast_f32_bf16<<<288, 256, 0, stream>>>((const float4*)w_qkv, wqkv_bf,
                                         3 * NH * DH * HIDN / 4);
  cast_f32_bf16<<<162, 256, 0, stream>>>((const float4*)w_o, wo_bf,
                                         HIDN * HIDN / 4);

  // QKV projection: [4096 x 1152] x [3456 x 1152]^T
  gemm_bf16_wmma<<<dim3(54, 64), 256, 0, stream>>>(hidden_bf, wqkv_bf, b_qkv,
                                                   qkv, 4096, 3456, HIDN);
  // RMSNorm + scale + RoPE + layout
  norm_rope<<<256, 256, 0, stream>>>(qkv, cosv, sinv, q_scale, k_scale,
                                     qpad, kpad, vtr);
  // Flash attention
  attn_kernel<<<dim3(16, 64), 128, 0, stream>>>(qpad, kpad, vtr, aout);
  // Output projection: [4096 x 1152] x [1152 x 1152]^T
  gemm_bf16_wmma<<<dim3(18, 64), 256, 0, stream>>>(aout, wo_bf, b_o,
                                                   out, 4096, HIDN, HIDN);
}
